// HybridBlock_67379446940104
// MI455X (gfx1250) — compile-verified, matrix operand order checked by reading, not run on hardware
//
#include <hip/hip_runtime.h>
#include <hip/hip_bf16.h>

#define B_  4
#define T_  2048
#define H_  1024
#define I_  2048
#define NS  16
#define KC  4
#define BT  (B_ * T_)     // 8192
#define TWO_I (2 * I_)    // 4096
#define PDIM 33

typedef __bf16 bf16_t;
typedef bf16_t v16bf __attribute__((ext_vector_type(16)));
typedef float  v8f   __attribute__((ext_vector_type(8)));
typedef int    v4i   __attribute__((ext_vector_type(4)));

// ---- feature detection: gfx1250 async global->LDS path ----
#if defined(__has_builtin)
#if __has_builtin(__builtin_amdgcn_global_load_async_to_lds_b128) && \
    __has_builtin(__builtin_amdgcn_s_wait_asynccnt)
#define USE_ASYNC_LDS 1
#endif
#endif
#ifndef USE_ASYNC_LDS
#define USE_ASYNC_LDS 0
#endif

// Builtin takes int4 pointers: AS1 (__device__) src, AS3 (LDS) dst.
// Generic LDS ptr low 32 bits == LDS byte offset (ISA 10.2), so go via integer.
#define AS1_V4I(p) ((__attribute__((address_space(1))) v4i*)(uintptr_t)(p))
#define AS3_V4I(p) ((__attribute__((address_space(3))) v4i*)(unsigned)(uintptr_t)(p))

// ---------- helpers ----------

__device__ inline bf16_t to_bf16(float f) {
    union { float f; unsigned u; } v; v.f = f;
    unsigned r = v.u + 0x7FFFu + ((v.u >> 16) & 1u);   // round-to-nearest-even
    union { unsigned short s; bf16_t b; } o;
    o.s = (unsigned short)(r >> 16);
    return o.b;
}

__device__ inline float silu(float x) { return x / (1.f + __expf(-x)); }

// 16x32 bf16 fragment from a row-major matrix (A-layout; B uses same packing on W^T rows).
// lane<16 : row = row0+lane,    K chunks [k0..k0+7]  and [k0+16..k0+23]
// lane>=16: row = row0+lane-16, K chunks [k0+8..k0+15] and [k0+24..k0+31]
__device__ inline v16bf load_frag(const bf16_t* __restrict__ base, int ld,
                                  int lane, int row0, int k0) {
    int l  = lane & 15;
    int hi = lane >> 4;
    const bf16_t* p = base + (size_t)(row0 + l) * ld + k0 + hi * 8;
    union { uint4 u[2]; v16bf v; } tmp;
    tmp.u[0] = *(const uint4*)(p);
    tmp.u[1] = *(const uint4*)(p + 16);
    return tmp.v;
}

// Same fragment gather from an LDS slab with leading dimension 32.
__device__ inline v16bf lds_frag(const bf16_t* base, int lane, int row0) {
    int l  = lane & 15;
    int hi = lane >> 4;
    const bf16_t* p = base + (row0 + l) * 32 + hi * 8;
    union { uint4 u[2]; v16bf v; } tmp;
    tmp.u[0] = *(const uint4*)(p);
    tmp.u[1] = *(const uint4*)(p + 16);
    return tmp.v;
}

// ---------- kernels ----------

__global__ __launch_bounds__(256)
void k_f32_to_bf16(const float* __restrict__ in, bf16_t* __restrict__ out, int n) {
    int i = blockIdx.x * 256 + threadIdx.x;
    if (i < n) out[i] = to_bf16(in[i]);
}

__global__ __launch_bounds__(256)
void k_layernorm(const float* __restrict__ x, const float* __restrict__ w,
                 const float* __restrict__ b, bf16_t* __restrict__ h) {
    __shared__ float red[512];
    int row = blockIdx.x;
    const float* xr = x + (size_t)row * H_;
    float s = 0.f, ss = 0.f;
    for (int i = threadIdx.x; i < H_; i += 256) {
        float v = xr[i]; s += v; ss += v * v;
    }
    red[threadIdx.x] = s; red[256 + threadIdx.x] = ss;
    __syncthreads();
    for (int off = 128; off > 0; off >>= 1) {
        if (threadIdx.x < off) {
            red[threadIdx.x]       += red[threadIdx.x + off];
            red[256 + threadIdx.x] += red[256 + threadIdx.x + off];
        }
        __syncthreads();
    }
    float mu  = red[0]   * (1.f / H_);
    float var = red[256] * (1.f / H_) - mu * mu;
    float inv = rsqrtf(var + 1e-5f);
    bf16_t* hr = h + (size_t)row * H_;
    for (int i = threadIdx.x; i < H_; i += 256)
        hr[i] = to_bf16((xr[i] - mu) * inv * w[i] + b[i]);
}

// C[M,N] = A[M,K] * Bw[N,K]^T (+ resid), bf16 inputs, f32 accumulate.
// Block: 8 waves (256 thr) -> 128(M) x 128(N) tile.
// Wave: 32x64 tile = 2 A-frags x 4 B-frags -> 8 WMMAs per 32-wide K slab.
// Triple-buffered async global->LDS staging, one barrier per slab.
__global__ __launch_bounds__(256)
void k_gemm_bf16(const bf16_t* __restrict__ A, const bf16_t* __restrict__ Bw,
                 const float* __restrict__ resid, float* __restrict__ C,
                 int M, int N, int Kd) {
    const int lane = threadIdx.x & 31;
    const int wave = threadIdx.x >> 5;
    const int wm = wave >> 1;                 // 0..3 -> M offset wm*32
    const int wn = wave & 1;                  // 0..1 -> N offset wn*64
    const int bm = blockIdx.y * 128;
    const int bn = blockIdx.x * 128;
    const int tmw = bm + wm * 32;
    const int tnw = bn + wn * 64;

    v8f acc[2][4] = {};

#if USE_ASYNC_LDS
    __shared__ __align__(16) bf16_t As[3][128][32];
    __shared__ __align__(16) bf16_t Bs[3][128][32];

    const int tid = threadIdx.x;

    // 1024 b128 chunks per slab (A:512 + B:512); 4 per thread (2 A + 2 B).
    auto copy_slab = [&](int k0, int buf) {
        #pragma unroll
        for (int h = 0; h < 2; ++h) {
            const int c   = tid + h * 256;
            const int row = c >> 2;
            const int cc  = (c & 3) * 8;
            __builtin_amdgcn_global_load_async_to_lds_b128(
                AS1_V4I(A  + (size_t)(bm + row) * Kd + k0 + cc),
                AS3_V4I(&As[buf][row][cc]), 0, 0);
            __builtin_amdgcn_global_load_async_to_lds_b128(
                AS1_V4I(Bw + (size_t)(bn + row) * Kd + k0 + cc),
                AS3_V4I(&Bs[buf][row][cc]), 0, 0);
        }
    };

    const int nk = Kd >> 5;
    copy_slab(0, 0);
    int cur = 0, nxt = 1;
    for (int ks = 0; ks < nk; ++ks) {
        if (ks + 1 < nk) {
            copy_slab((ks + 1) << 5, nxt);        // prefetch next slab
            __builtin_amdgcn_s_wait_asynccnt(4);  // drain only current slab
        } else {
            __builtin_amdgcn_s_wait_asynccnt(0);
        }
        __syncthreads();   // slab `cur` visible; also fences WAR for buffer cur+1 (read at ks-2)

        v16bf a0 = lds_frag(&As[cur][0][0], lane, wm * 32);
        v16bf a1 = lds_frag(&As[cur][0][0], lane, wm * 32 + 16);
        v16bf b0 = lds_frag(&Bs[cur][0][0], lane, wn * 64);
        v16bf b1 = lds_frag(&Bs[cur][0][0], lane, wn * 64 + 16);
        v16bf b2 = lds_frag(&Bs[cur][0][0], lane, wn * 64 + 32);
        v16bf b3 = lds_frag(&Bs[cur][0][0], lane, wn * 64 + 48);

        acc[0][0] = __builtin_amdgcn_wmma_f32_16x16x32_bf16(false, a0, false, b0, (short)0, acc[0][0], false, false);
        acc[0][1] = __builtin_amdgcn_wmma_f32_16x16x32_bf16(false, a0, false, b1, (short)0, acc[0][1], false, false);
        acc[0][2] = __builtin_amdgcn_wmma_f32_16x16x32_bf16(false, a0, false, b2, (short)0, acc[0][2], false, false);
        acc[0][3] = __builtin_amdgcn_wmma_f32_16x16x32_bf16(false, a0, false, b3, (short)0, acc[0][3], false, false);
        acc[1][0] = __builtin_amdgcn_wmma_f32_16x16x32_bf16(false, a1, false, b0, (short)0, acc[1][0], false, false);
        acc[1][1] = __builtin_amdgcn_wmma_f32_16x16x32_bf16(false, a1, false, b1, (short)0, acc[1][1], false, false);
        acc[1][2] = __builtin_amdgcn_wmma_f32_16x16x32_bf16(false, a1, false, b2, (short)0, acc[1][2], false, false);
        acc[1][3] = __builtin_amdgcn_wmma_f32_16x16x32_bf16(false, a1, false, b3, (short)0, acc[1][3], false, false);

        cur = nxt;
        nxt = (nxt == 2) ? 0 : nxt + 1;
    }
#else
    // ---- fallback: direct global-load pipeline (dead on async-capable toolchains) ----
    for (int k0 = 0; k0 < Kd; k0 += 32) {
        v16bf a0 = load_frag(A,  Kd, lane, tmw,      k0);
        v16bf a1 = load_frag(A,  Kd, lane, tmw + 16, k0);
        v16bf b0 = load_frag(Bw, Kd, lane, tnw,      k0);
        v16bf b1 = load_frag(Bw, Kd, lane, tnw + 16, k0);
        v16bf b2 = load_frag(Bw, Kd, lane, tnw + 32, k0);
        v16bf b3 = load_frag(Bw, Kd, lane, tnw + 48, k0);
        acc[0][0] = __builtin_amdgcn_wmma_f32_16x16x32_bf16(false, a0, false, b0, (short)0, acc[0][0], false, false);
        acc[0][1] = __builtin_amdgcn_wmma_f32_16x16x32_bf16(false, a0, false, b1, (short)0, acc[0][1], false, false);
        acc[0][2] = __builtin_amdgcn_wmma_f32_16x16x32_bf16(false, a0, false, b2, (short)0, acc[0][2], false, false);
        acc[0][3] = __builtin_amdgcn_wmma_f32_16x16x32_bf16(false, a0, false, b3, (short)0, acc[0][3], false, false);
        acc[1][0] = __builtin_amdgcn_wmma_f32_16x16x32_bf16(false, a1, false, b0, (short)0, acc[1][0], false, false);
        acc[1][1] = __builtin_amdgcn_wmma_f32_16x16x32_bf16(false, a1, false, b1, (short)0, acc[1][1], false, false);
        acc[1][2] = __builtin_amdgcn_wmma_f32_16x16x32_bf16(false, a1, false, b2, (short)0, acc[1][2], false, false);
        acc[1][3] = __builtin_amdgcn_wmma_f32_16x16x32_bf16(false, a1, false, b3, (short)0, acc[1][3], false, false);
    }
#endif

    // C/D layout: VGPR v -> row v + 8*(lane>=16), col = lane&15
    const int rbase = (lane >> 4) << 3;
    const int col   = lane & 15;
    #pragma unroll
    for (int am = 0; am < 2; ++am) {
        #pragma unroll
        for (int v = 0; v < 8; ++v) {
            const int row = tmw + am * 16 + rbase + v;
            const size_t o = (size_t)row * N + tnw + col;
            if (resid) {
                C[o]      = resid[o]      + acc[am][0][v];
                C[o + 16] = resid[o + 16] + acc[am][1][v];
                C[o + 32] = resid[o + 32] + acc[am][2][v];
                C[o + 48] = resid[o + 48] + acc[am][3][v];
            } else {
                C[o]      = acc[am][0][v];
                C[o + 16] = acc[am][1][v];
                C[o + 32] = acc[am][2][v];
                C[o + 48] = acc[am][3][v];
            }
        }
    }
}

__global__ __launch_bounds__(256)
void k_conv_silu(const float* __restrict__ xz, const float* __restrict__ conv_w,
                 float* __restrict__ xc) {
    int idx = blockIdx.x * 256 + threadIdx.x;
    if (idx >= BT * I_) return;
    int i  = idx % I_;
    int bt = idx / I_;
    int t  = bt % T_;
    float acc = 0.f;
    #pragma unroll
    for (int k = 0; k < KC; ++k) {
        int tp = t + k - (KC - 1);
        if (tp >= 0)
            acc = fmaf(conv_w[i * KC + k],
                       xz[(size_t)(bt + k - (KC - 1)) * TWO_I + i], acc);
    }
    xc[idx] = silu(acc);
}

__global__ __launch_bounds__(256)
void k_proj(const float* __restrict__ xc, const float* __restrict__ W_xp,
            float* __restrict__ proj) {
    int idx = blockIdx.x * 256 + threadIdx.x;
    if (idx >= BT * PDIM) return;
    int bt = idx / PDIM, p = idx % PDIM;
    const float* xr = xc   + (size_t)bt * I_;
    const float* wr = W_xp + (size_t)p  * I_;
    float acc = 0.f;
    #pragma unroll 4
    for (int k = 0; k < I_; ++k) acc = fmaf(xr[k], wr[k], acc);
    proj[idx] = acc;
}

// Selective scan: lane (n = lane&15) owns one state of one (b,i) channel.
__global__ __launch_bounds__(256)
void k_scan(const float* __restrict__ proj, const float* __restrict__ xc,
            const float* __restrict__ st_in, const float* __restrict__ A_log,
            float* __restrict__ y, float* __restrict__ st_out) {
    const int n   = threadIdx.x & 15;
    const int grp = threadIdx.x >> 4;           // 16 channels per block
    const int ci  = blockIdx.x * 16 + grp;      // 0..B*I-1
    const int b   = ci >> 11;                   // I_ = 2048
    const int i   = ci & (I_ - 1);

    const float Aneg = -__expf(A_log[i * NS + n]);
    float s = st_in[((size_t)b * I_ + i) * NS + n];

    const float* projb = proj + (size_t)b * T_ * PDIM;
    const float* xcb   = xc   + ((size_t)b * T_) * I_ + i;
    float*       yb    = y    + ((size_t)b * T_) * I_ + i;

    for (int t = 0; t < T_; ++t) {
        const float* pr = projb + (size_t)t * PDIM;
        float draw  = pr[0];
        float delta = (draw > 20.f) ? draw : log1pf(__expf(draw));
        float bt = pr[1 + n];
        float ct = pr[1 + NS + n];
        float xt = xcb[(size_t)t * I_];
        s = fmaf(s, __expf(delta * Aneg), xt * bt);
        float p = s * ct;
        p += __shfl_xor(p, 1);
        p += __shfl_xor(p, 2);
        p += __shfl_xor(p, 4);
        p += __shfl_xor(p, 8);
        if (n == 0) yb[(size_t)t * I_] = p;
    }
    st_out[((size_t)b * I_ + i) * NS + n] = s;
}

__global__ __launch_bounds__(256)
void k_gate(const float* __restrict__ y, const float* __restrict__ xc,
            const float* __restrict__ xz, const float* __restrict__ Dv,
            bf16_t* __restrict__ ybf) {
    int idx = blockIdx.x * 256 + threadIdx.x;
    if (idx >= BT * I_) return;
    int i  = idx % I_;
    int bt = idx / I_;
    float z  = xz[(size_t)bt * TWO_I + I_ + i];
    float yy = (y[idx] + xc[idx] * Dv[i]) * silu(z);
    ybf[idx] = to_bf16(yy);
}

// ---------- launch ----------

extern "C" void kernel_launch(void* const* d_in, const int* in_sizes, int n_in,
                              void* d_out, int out_size, void* d_ws, size_t ws_size,
                              hipStream_t stream) {
    (void)in_sizes; (void)n_in; (void)out_size; (void)ws_size;
    const float* x      = (const float*)d_in[0];
    const float* st_in  = (const float*)d_in[1];
    const float* ln_w   = (const float*)d_in[2];
    const float* ln_b   = (const float*)d_in[3];
    const float* W_in   = (const float*)d_in[4];
    const float* conv_w = (const float*)d_in[5];
    const float* W_xp   = (const float*)d_in[6];
    const float* A_log  = (const float*)d_in[7];
    const float* Dv     = (const float*)d_in[8];
    const float* W_out  = (const float*)d_in[9];

    float* out       = (float*)d_out;                 // (B,T,H)
    float* state_out = out + (size_t)BT * H_;         // (B,I,N)

    char* ws = (char*)d_ws;
    bf16_t* h_bf    = (bf16_t*)(ws + 0ull);            // 16 MB
    bf16_t* win_bf  = (bf16_t*)(ws + 16777216ull);     //  8 MB
    bf16_t* wout_bf = (bf16_t*)(ws + 25165824ull);     //  4 MB
    float*  xz      = (float*) (ws + 29360128ull);     // 128 MB
    float*  xc      = (float*) (ws + 163577856ull);    // 64 MB
    float*  proj    = (float*) (ws + 230686720ull);    // ~1 MB
    float*  y       = (float*) (ws + 231768064ull);    // 64 MB
    bf16_t* y_bf    = (bf16_t*)(ws + 298876928ull);    // 32 MB

    k_f32_to_bf16<<<(TWO_I * H_) / 256, 256, 0, stream>>>(W_in, win_bf, TWO_I * H_);
    k_f32_to_bf16<<<(H_ * I_) / 256, 256, 0, stream>>>(W_out, wout_bf, H_ * I_);
    k_layernorm<<<BT, 256, 0, stream>>>(x, ln_w, ln_b, h_bf);

    dim3 g1(TWO_I / 128, BT / 128);   // (32, 64)
    k_gemm_bf16<<<g1, 256, 0, stream>>>(h_bf, win_bf, nullptr, xz, BT, TWO_I, H_);

    k_conv_silu<<<(BT * I_) / 256, 256, 0, stream>>>(xz, conv_w, xc);
    k_proj<<<(BT * PDIM + 255) / 256, 256, 0, stream>>>(xc, W_xp, proj);
    k_scan<<<(B_ * I_) / 16, 256, 0, stream>>>(proj, xc, st_in, A_log, y, state_out);
    k_gate<<<(BT * I_) / 256, 256, 0, stream>>>(y, xc, xz, Dv, y_bf);

    dim3 g2(H_ / 128, BT / 128);      // (8, 64)
    k_gemm_bf16<<<g2, 256, 0, stream>>>(y_bf, wout_bf, x, out, BT, H_, I_);
}
